// CrossAttention_61229053772008
// MI455X (gfx1250) — matrix operand order repaired and checked by reading, not compiled, not run on hardware
//
#include <hip/hip_runtime.h>
#include <hip/hip_bf16.h>

// ---------------------------------------------------------------------------
// Fused: LayerNorm(x) @ Wq^T, LayerNorm(xf) @ Wk^T / Wv^T, per-head gate,
//        y1 = (1-w)*q, y2 = w*v.   B*T = 32768 rows, D=512, L=768, Hd=512, H=8
// GEMMs in bf16 via v_wmma_f32_16x16x32_bf16 (fp32 accumulate).
// ---------------------------------------------------------------------------

typedef __attribute__((ext_vector_type(16))) __bf16 v16bf;
typedef __attribute__((ext_vector_type(8)))  float  v8f;

#define DK 512          // x feature dim (K for q GEMM)
#define LK 768          // xf feature dim (K for k/v GEMMs)
#define HD 512          // output feature dim (Hd)
#define ROWS_PB 64      // rows per workgroup
#define NWQ  (512*512)  // Wq elements
#define NWK  (512*768)  // Wk elements
#define NWV  (512*768)  // Wv elements

__device__ __forceinline__ unsigned short f2bf(float f) {
    unsigned u = __float_as_uint(f);
    u += 0x7fffu + ((u >> 16) & 1u);       // round-to-nearest-even
    return (unsigned short)(u >> 16);
}

union FragBF {
    uint4 u[2];   // 32 bytes
    v16bf v;      // 16 bf16 = 8 VGPRs
};

// ---------------------------------------------------------------------------
// Prologue: convert Wq|Wk|Wv fp32 -> bf16 into workspace (read once, L2-hot).
// ---------------------------------------------------------------------------
__global__ void cvt_weights_kernel(const float* __restrict__ wq,
                                   const float* __restrict__ wk,
                                   const float* __restrict__ wv,
                                   unsigned short* __restrict__ out) {
    int i = blockIdx.x * 256 + threadIdx.x;
    if (i < NWQ)                    out[i] = f2bf(wq[i]);
    else if (i < NWQ + NWK)         out[i] = f2bf(wk[i - NWQ]);
    else if (i < NWQ + NWK + NWV)   out[i] = f2bf(wv[i - NWQ - NWK]);
}

// ---------------------------------------------------------------------------
// 64x64 GEMM for one wave: C[64x64] += A[64xK] (LDS bf16, row-major)
//                                    x B[64xK]^T (global bf16, [N,K] row-major)
// ---------------------------------------------------------------------------
template<int KSTEPS, int KDIM>
__device__ __forceinline__ void gemm64(const unsigned short* __restrict__ ldsA,
                                       const unsigned short* __restrict__ wB,
                                       int wv, int lane, v8f acc[16]) {
    const int hi  = lane >> 4;   // 0: K-low chunks, 1: K-high chunks
    const int l16 = lane & 15;
#pragma unroll 1
    for (int ks = 0; ks < KSTEPS; ++ks) {
        FragBF a[4], b[4];
#pragma unroll
        for (int mt = 0; mt < 4; ++mt) {
            // A 16x32 bf16 fragment: lanes 0-15 row M=l16 hold K {0..7,16..23},
            // lanes 16-31 same rows hold K {8..15,24..31}.
            const unsigned short* p = ldsA + (mt * 16 + l16) * KDIM + ks * 32 + hi * 8;
            a[mt].u[0] = *(const uint4*)(p);        // K base + {0..7}
            a[mt].u[1] = *(const uint4*)(p + 16);   // K base + {16..23}
        }
#pragma unroll
        for (int nt = 0; nt < 4; ++nt) {
            // B 32x16 bf16 fragment: lane = column n, lanes 0-15 K {0..7,16..23},
            // lanes 16-31 K {8..15,24..31} -> two 16B reads per lane.
            const unsigned short* p = wB + (unsigned)(wv * 64 + nt * 16 + l16) * KDIM
                                         + ks * 32 + hi * 8;
            b[nt].u[0] = *(const uint4*)(p);
            b[nt].u[1] = *(const uint4*)(p + 16);
        }
#pragma unroll
        for (int mt = 0; mt < 4; ++mt)
#pragma unroll
            for (int nt = 0; nt < 4; ++nt)
                acc[mt * 4 + nt] = __builtin_amdgcn_wmma_f32_16x16x32_bf16(
                    false, a[mt].v, false, b[nt].v,
                    (short)0, acc[mt * 4 + nt], false, false);
    }
}

// ---------------------------------------------------------------------------
// Main fused kernel. 256 threads = 8 waves; wave w owns head w (64 columns).
// ---------------------------------------------------------------------------
__global__ __launch_bounds__(256, 1)
void fused_xattn_kernel(const float* __restrict__ x,
                        const float* __restrict__ xf,
                        const float* __restrict__ nw, const float* __restrict__ nb,
                        const float* __restrict__ tw, const float* __restrict__ tb,
                        const unsigned short* __restrict__ wbf,  // Wq|Wk|Wv bf16
                        float* __restrict__ out, int rows) {
    extern __shared__ unsigned short lds[];           // 64*512 + 64*768 bf16
    unsigned short* ldsX = lds;                       // LN(x)  tile, 64x512
    unsigned short* ldsF = lds + ROWS_PB * DK;        // LN(xf) tile, 64x768

    const int wv   = threadIdx.x >> 5;
    const int lane = threadIdx.x & 31;
    const long row0 = (long)blockIdx.x * ROWS_PB;

    // ---------------- Phase 1: LayerNorm -> bf16 tiles in LDS --------------
    // x: 512 cols -> 16 floats/lane; 8 rows per wave.
#pragma unroll 1
    for (int j = 0; j < 8; ++j) {
        const int rl = wv * 8 + j;
        const float4* xr = (const float4*)(x + (row0 + rl) * DK);
        float vals[16];
        float s = 0.f, ss = 0.f;
#pragma unroll
        for (int c = 0; c < 4; ++c) {
            float4 t = xr[lane * 4 + c];
            vals[c*4+0] = t.x; vals[c*4+1] = t.y; vals[c*4+2] = t.z; vals[c*4+3] = t.w;
            s  += t.x + t.y + t.z + t.w;
            ss += t.x*t.x + t.y*t.y + t.z*t.z + t.w*t.w;
        }
#pragma unroll
        for (int m = 16; m >= 1; m >>= 1) {
            s  += __shfl_xor(s,  m, 32);
            ss += __shfl_xor(ss, m, 32);
        }
        const float mean = s * (1.f / DK);
        const float rstd = rsqrtf(ss * (1.f / DK) - mean * mean + 1e-5f);
        unsigned int* dst = (unsigned int*)(ldsX + rl * DK + lane * 16);
#pragma unroll
        for (int e = 0; e < 16; e += 2) {
            const int col = lane * 16 + e;
            float a0 = (vals[e]   - mean) * rstd * nw[col]     + nb[col];
            float a1 = (vals[e+1] - mean) * rstd * nw[col + 1] + nb[col + 1];
            dst[e >> 1] = (unsigned)f2bf(a0) | ((unsigned)f2bf(a1) << 16);
        }
    }
    // xf: 768 cols -> 24 floats/lane; 8 rows per wave.
#pragma unroll 1
    for (int j = 0; j < 8; ++j) {
        const int rl = wv * 8 + j;
        const float4* xr = (const float4*)(xf + (row0 + rl) * LK);
        float vals[24];
        float s = 0.f, ss = 0.f;
#pragma unroll
        for (int c = 0; c < 6; ++c) {
            float4 t = xr[lane * 6 + c];
            vals[c*4+0] = t.x; vals[c*4+1] = t.y; vals[c*4+2] = t.z; vals[c*4+3] = t.w;
            s  += t.x + t.y + t.z + t.w;
            ss += t.x*t.x + t.y*t.y + t.z*t.z + t.w*t.w;
        }
#pragma unroll
        for (int m = 16; m >= 1; m >>= 1) {
            s  += __shfl_xor(s,  m, 32);
            ss += __shfl_xor(ss, m, 32);
        }
        const float mean = s * (1.f / LK);
        const float rstd = rsqrtf(ss * (1.f / LK) - mean * mean + 1e-5f);
        unsigned int* dst = (unsigned int*)(ldsF + rl * LK + lane * 24);
#pragma unroll
        for (int e = 0; e < 24; e += 2) {
            const int col = lane * 24 + e;
            float a0 = (vals[e]   - mean) * rstd * tw[col]     + tb[col];
            float a1 = (vals[e+1] - mean) * rstd * tw[col + 1] + tb[col + 1];
            dst[e >> 1] = (unsigned)f2bf(a0) | ((unsigned)f2bf(a1) << 16);
        }
    }
    __syncthreads();

    const unsigned short* wq_b = wbf;
    const unsigned short* wk_b = wbf + NWQ;
    const unsigned short* wv_b = wbf + NWQ + NWK;

    // ---------------- Phase 2: q and k GEMMs (each wave: 64x64) ------------
    v8f accQ[16], accK[16];
#pragma unroll
    for (int t = 0; t < 16; ++t) { accQ[t] = (v8f)0.f; accK[t] = (v8f)0.f; }
    gemm64<DK / 32, DK>(ldsX, wq_b, wv, lane, accQ);
    gemm64<LK / 32, LK>(ldsF, wk_b, wv, lane, accK);

    // ---------------- gate: w[row] = (q . k)/sqrt(64) per head -------------
    // C layout: VGPR i, lanes 0-15 -> row mt*16+i, lanes 16-31 -> row mt*16+i+8;
    // column = lane&15. Width-16 butterfly stays within a row group.
    float gate[4][8];
#pragma unroll
    for (int mt = 0; mt < 4; ++mt)
#pragma unroll
        for (int i = 0; i < 8; ++i) {
            float p = 0.f;
#pragma unroll
            for (int nt = 0; nt < 4; ++nt)
                p += accQ[mt * 4 + nt][i] * accK[mt * 4 + nt][i];
#pragma unroll
            for (int m = 8; m >= 1; m >>= 1) p += __shfl_xor(p, m, 16);
            gate[mt][i] = p * 0.125f;   // 1/sqrt(64)
        }

    // ---------------- y1 = (1-w) * q ---------------------------------------
    const int hi  = lane >> 4;
    const int l16 = lane & 15;
    float* __restrict__ y1 = out;
    float* __restrict__ y2 = out + (size_t)rows * HD;
#pragma unroll
    for (int mt = 0; mt < 4; ++mt)
#pragma unroll
        for (int nt = 0; nt < 4; ++nt)
#pragma unroll
            for (int i = 0; i < 8; ++i) {
                const int  rl  = mt * 16 + i + hi * 8;
                const int  col = wv * 64 + nt * 16 + l16;
                y1[(row0 + rl) * HD + col] = (1.f - gate[mt][i]) * accQ[mt * 4 + nt][i];
            }

    // ---------------- v GEMM (reuse accK registers), y2 = w * v ------------
#pragma unroll
    for (int t = 0; t < 16; ++t) accK[t] = (v8f)0.f;
    gemm64<LK / 32, LK>(ldsF, wv_b, wv, lane, accK);
#pragma unroll
    for (int mt = 0; mt < 4; ++mt)
#pragma unroll
        for (int nt = 0; nt < 4; ++nt)
#pragma unroll
            for (int i = 0; i < 8; ++i) {
                const int  rl  = mt * 16 + i + hi * 8;
                const int  col = wv * 64 + nt * 16 + l16;
                y2[(row0 + rl) * HD + col] = gate[mt][i] * accK[mt * 4 + nt][i];
            }
}

// ---------------------------------------------------------------------------
extern "C" void kernel_launch(void* const* d_in, const int* in_sizes, int n_in,
                              void* d_out, int out_size, void* d_ws, size_t ws_size,
                              hipStream_t stream) {
    const float* x   = (const float*)d_in[0];
    const float* xf  = (const float*)d_in[1];
    const float* nw  = (const float*)d_in[2];
    const float* nb  = (const float*)d_in[3];
    const float* tw  = (const float*)d_in[4];
    const float* tb  = (const float*)d_in[5];
    const float* Wq  = (const float*)d_in[6];
    const float* Wk  = (const float*)d_in[7];
    const float* Wv  = (const float*)d_in[8];
    float* out = (float*)d_out;

    const int rows = in_sizes[0] / DK;          // B*T = 32768
    unsigned short* wbf = (unsigned short*)d_ws;

    // 1) weights fp32 -> bf16 (L2-resident afterwards)
    const int nwt = NWQ + NWK + NWV;            // 1,048,576 elements
    cvt_weights_kernel<<<(nwt + 255) / 256, 256, 0, stream>>>(Wq, Wk, Wv, wbf);

    // 2) fused LN + 3 GEMMs + gate + elementwise
    const size_t lds_bytes = (size_t)(ROWS_PB * DK + ROWS_PB * LK) * sizeof(unsigned short); // 160 KB
    fused_xattn_kernel<<<rows / ROWS_PB, 256, lds_bytes, stream>>>(
        x, xf, nw, nb, tw, tb, wbf, out, rows);
}
